// MaskedMultiScaleSegHead_64201171141228
// MI455X (gfx1250) — matrix operand order repaired and checked by reading, not compiled, not run on hardware
//
#include <hip/hip_runtime.h>
#include <hip/hip_bf16.h>
#include <math.h>

// ---------------------------------------------------------------------------
// MaskedMultiScaleSegHead for MI455X (gfx1250), wave32 + WMMA bf16.
//   * pool-before-conv (1x1x1 conv commutes with block-average pooling)
//   * mask path folded:  prev = (me @ mask_w) @ feat + (me . mask_b)
// GEMMs: v_wmma_f32_16x16x32_bf16, 8 waves/block, 32x256 block tile,
// 16x64 per wave (4 WMMA per A-fragment), double-buffered LDS (1 barrier/step),
// float4 global loads, bf16 LDS tiles with aligned ds_load_b128 fragments.
// ---------------------------------------------------------------------------

#define BB   2
#define QQ   32
#define HH   256
#define NHD  8
#define NTOK 4096
#define CCLS 30

typedef __attribute__((ext_vector_type(16))) __bf16 v16bf;
typedef __attribute__((ext_vector_type(8)))  __bf16 v8bf;
typedef __attribute__((ext_vector_type(4)))  __bf16 v4bf;
typedef __attribute__((ext_vector_type(8)))  float  v8f;

// Plain cast: lets ISel select the native f32->bf16 convert (packed when it
// can fuse adjacent casts) instead of a hand-rolled RNE bit sequence.
__device__ __forceinline__ __bf16 f2bf(float f) { return (__bf16)f; }

__device__ __forceinline__ float gelu_exact(float x) {
  return 0.5f * x * (1.0f + erff(x * 0.70710678118654752f));
}
__device__ __forceinline__ float sigmoidf(float x) {
  return 1.0f / (1.0f + __expf(-x));
}

// ---------------------------------------------------------------------------
// WMMA GEMM:  OUT[b][m][n] = sum_k A[b][m][k] * X[b][k][n] + bias[b][m]
// A row-major [Mtot][K] f32, X row-major [K][N] f32, per-operand batch strides
// (0 => broadcast over batch). Requires Mtot%32==0, N%256==0, K%32==0.
// Block: 256 threads = 8 waves; tile 32(M) x 256(N); wave tile 16x64.
// ---------------------------------------------------------------------------
__global__ __launch_bounds__(256) void wmma_gemm_bf16(
    const float* __restrict__ A, const float* __restrict__ X,
    const float* __restrict__ bias, float* __restrict__ OUT,
    int Mtot, int N, int K,
    long aBS, long xBS, long bBS, long oBS)
{
  __shared__ __align__(16) __bf16 Alds[2][32][40];    // [m][k], 80B rows
  __shared__ __align__(16) __bf16 Blds[2][256][40];   // [n][k] (transposed)

  const int b  = blockIdx.z;
  const int m0 = blockIdx.y * 32;
  const int n0 = blockIdx.x * 256;
  A += (long)b * aBS;
  X += (long)b * xBS;
  OUT += (long)b * oBS;
  const float* bp = bias ? bias + (long)b * bBS : nullptr;

  const int tid  = threadIdx.x;
  const int wave = tid >> 5;
  const int lane = tid & 31;
  const int wm   = (wave >> 2) << 4;   // 0 | 16
  const int wnb  = (wave & 3) << 6;    // 0,64,128,192
  const int half = lane >> 4;
  const int l16  = lane & 15;

  // per-thread tile-load coordinates
  const int am  = tid >> 3;            // A: 32 rows x 8 float4
  const int ak4 = (tid & 7) << 2;
  const int bkr = tid >> 6;            // B: k rows {bkr, bkr+4, ...}
  const int bn  = (tid & 63) << 2;     // B: element column (float4)

  v8f acc[4] = {};
  float4 aR;
  float4 bR[8];

  auto loadTiles = [&](int k0) {
    aR = *(const float4*)&A[(long)(m0 + am) * K + (k0 + ak4)];
    #pragma unroll
    for (int i = 0; i < 8; i++) {
      int k = bkr + i * 4;
      bR[i] = *(const float4*)&X[(long)(k0 + k) * N + (n0 + bn)];
    }
  };
  auto storeTiles = [&](int buf) {
    v4bf ap;
    ap[0] = f2bf(aR.x); ap[1] = f2bf(aR.y);
    ap[2] = f2bf(aR.z); ap[3] = f2bf(aR.w);
    *(v4bf*)&Alds[buf][am][ak4] = ap;          // ds_store_b64
    #pragma unroll
    for (int i = 0; i < 8; i++) {
      int k = bkr + i * 4;
      Blds[buf][bn + 0][k] = f2bf(bR[i].x);    // transpose on store
      Blds[buf][bn + 1][k] = f2bf(bR[i].y);
      Blds[buf][bn + 2][k] = f2bf(bR[i].z);
      Blds[buf][bn + 3][k] = f2bf(bR[i].w);
    }
  };
  auto compute = [&](int buf) {
    // A fragment (ISA 16-bit A 16x32 layout): lane half -> K {0..7,16..23}
    // or {8..15,24..31}; two aligned ds_load_b128.
    v8bf alo = *(const v8bf*)&Alds[buf][wm + l16][half * 8];
    v8bf ahi = *(const v8bf*)&Alds[buf][wm + l16][half * 8 + 16];
    v16bf av;
    #pragma unroll
    for (int i = 0; i < 8; i++) { av[i] = alo[i]; av[8 + i] = ahi[i]; }
    #pragma unroll
    for (int j = 0; j < 4; j++) {
      // B fragment (16-bit B 32x16): lanes 0-15 K={0..7,16..23}, 16-31 K={8..15,24..31}.
      const __bf16* pb = &Blds[buf][wnb + (j << 4) + l16][half << 3];
      v8bf blo = *(const v8bf*)pb;
      v8bf bhi = *(const v8bf*)(pb + 16);
      v16bf bv;
      #pragma unroll
      for (int i = 0; i < 8; i++) { bv[i] = blo[i]; bv[8 + i] = bhi[i]; }
      acc[j] = __builtin_amdgcn_wmma_f32_16x16x32_bf16(
          false, av, false, bv, (short)0, acc[j], false, false);
    }
  };

  loadTiles(0);
  storeTiles(0);
  __syncthreads();
  int buf = 0;
  for (int k0 = 0; k0 < K; k0 += 32) {
    const bool hasNext = (k0 + 32) < K;
    if (hasNext) loadTiles(k0 + 32);                   // register-staged
    if (k0 + 64 < K)                                   // global_prefetch_b8
      __builtin_prefetch(&X[(long)(k0 + 64 + bkr) * N + n0 + bn], 0, 1);
    compute(buf);                                      // 4x WMMA
    if (hasNext) storeTiles(buf ^ 1);
    __syncthreads();                                   // single barrier/step
    buf ^= 1;
  }

  // C layout: lanes 0-15 -> N=lane, M=r; lanes 16-31 -> N=lane-16, M=r+8
  #pragma unroll
  for (int j = 0; j < 4; j++) {
    const int m = m0 + wm + half * 8;
    const int n = n0 + wnb + (j << 4) + l16;
    #pragma unroll
    for (int r = 0; r < 8; r++) {
      float v = acc[j][r];
      if (bp) v += bp[m + r];
      OUT[(long)(m + r) * N + n] = v;
    }
  }
}

// ---------------------------------------------------------------------------
// Block-average pool [B][C][S^3] -> [B][C][16^3]; f = S/16 (1,2,3)
// ---------------------------------------------------------------------------
__global__ void pool_kernel(const float* __restrict__ in, float* __restrict__ out,
                            int S, int f, int total)
{
  int idx = blockIdx.x * 256 + threadIdx.x;
  if (idx >= total) return;
  int x = idx & 15, y = (idx >> 4) & 15, z = (idx >> 8) & 15;
  int bc = idx >> 12;
  const float* p = in + (long)bc * S * S * S;
  float s = 0.f;
  for (int dz = 0; dz < f; dz++)
    for (int dy = 0; dy < f; dy++)
      for (int dx = 0; dx < f; dx++)
        s += p[((z * f + dz) * S + (y * f + dy)) * S + (x * f + dx)];
  out[idx] = s / (float)(f * f * f);
}

// ---------------------------------------------------------------------------
// Small dense: OUT[r][o] = act( sum_k X[r][k]*W + bias[o] )
// wT=1: W[o*K+k] (torch x@W.T);  wT=0: W[k*O+o] (x@W)
// ---------------------------------------------------------------------------
__global__ void linear_kernel(const float* __restrict__ X, const float* __restrict__ W,
                              const float* __restrict__ bias, float* __restrict__ OUT,
                              int rows, int K, int O, int wT, int act)
{
  int idx = blockIdx.x * 256 + threadIdx.x;
  if (idx >= rows * O) return;
  int r = idx / O, o = idx % O;
  const float* x = X + r * K;
  float acc = bias ? bias[o] : 0.f;
  if (wT) {
    const float* w = W + o * K;
    for (int k = 0; k < K; k++) acc += x[k] * w[k];
  } else {
    for (int k = 0; k < K; k++) acc += x[k] * W[k * O + o];
  }
  if (act) acc = gelu_exact(acc);
  OUT[idx] = acc;
}

__global__ void dot_rows_kernel(const float* __restrict__ X, const float* __restrict__ v,
                                float* __restrict__ out, int rows, int K)
{
  int r = blockIdx.x * blockDim.x + threadIdx.x;
  if (r >= rows) return;
  const float* x = X + r * K;
  float acc = 0.f;
  for (int k = 0; k < K; k++) acc += x[k] * v[k];
  out[r] = acc;
}

// ---------------------------------------------------------------------------
// out[row] = LayerNorm(A[row] + Badd[row]) * g + beta   (H=256, block=row)
// ---------------------------------------------------------------------------
__global__ __launch_bounds__(256) void add_ln_kernel(
    const float* __restrict__ A, const float* __restrict__ Badd,
    const float* __restrict__ g, const float* __restrict__ beta,
    float* __restrict__ out)
{
  __shared__ float red[256];
  int r = blockIdx.x, t = threadIdx.x;
  float v = A[r * 256 + t] + (Badd ? Badd[r * 256 + t] : 0.f);
  red[t] = v; __syncthreads();
  for (int s = 128; s > 0; s >>= 1) { if (t < s) red[t] += red[t + s]; __syncthreads(); }
  float mean = red[0] * (1.f / 256.f); __syncthreads();
  float d = v - mean;
  red[t] = d * d; __syncthreads();
  for (int s = 128; s > 0; s >>= 1) { if (t < s) red[t] += red[t + s]; __syncthreads(); }
  float var = red[0] * (1.f / 256.f);
  out[r * 256 + t] = d * rsqrtf(var + 1e-5f) * g[t] + beta[t];
}

// ---------------------------------------------------------------------------
// Cross attention, one block per (q, head, b). 4096 keys, softmax in LDS.
// Qp: [B][32][256] (already Wq-projected), KT/VT: [B][256][4096],
// biasA: [B][32][4096] or null. out: [B][32][256].
// ---------------------------------------------------------------------------
__global__ __launch_bounds__(256) void cross_attn_kernel(
    const float* __restrict__ Qp, const float* __restrict__ KT,
    const float* __restrict__ VT, const float* __restrict__ biasA,
    float* __restrict__ outp)
{
  __shared__ float s[NTOK];
  __shared__ float red[256];
  int qi = blockIdx.x, h = blockIdx.y, b = blockIdx.z, t = threadIdx.x;
  const float scale = 0.17677669529663689f;   // (H/NH)^-0.5 = 32^-0.5
  float qv[32];
  const float* qp = Qp + (b * QQ + qi) * HH + h * 32;
  #pragma unroll
  for (int d = 0; d < 32; d++) qv[d] = qp[d] * scale;
  const float* Kb = KT + (b * HH + h * 32) * NTOK;
  const float* Vb = VT + (b * HH + h * 32) * NTOK;
  const float* bb = biasA ? biasA + (b * QQ + qi) * NTOK : nullptr;

  float lmax = -1e30f;
  for (int n = t; n < NTOK; n += 256) {
    float acc = bb ? bb[n] : 0.f;
    #pragma unroll
    for (int d = 0; d < 32; d++) acc += qv[d] * Kb[d * NTOK + n];
    s[n] = acc; lmax = fmaxf(lmax, acc);
  }
  red[t] = lmax; __syncthreads();
  for (int st = 128; st > 0; st >>= 1) { if (t < st) red[t] = fmaxf(red[t], red[t + st]); __syncthreads(); }
  float M = red[0]; __syncthreads();

  float lsum = 0.f;
  for (int n = t; n < NTOK; n += 256) { float e = __expf(s[n] - M); s[n] = e; lsum += e; }
  red[t] = lsum; __syncthreads();
  for (int st = 128; st > 0; st >>= 1) { if (t < st) red[t] += red[t + st]; __syncthreads(); }
  float inv = 1.f / red[0]; __syncthreads();

  float o[32];
  #pragma unroll
  for (int d = 0; d < 32; d++) o[d] = 0.f;
  for (int n = t; n < NTOK; n += 256) {
    float p = s[n];
    #pragma unroll
    for (int d = 0; d < 32; d++) o[d] += p * Vb[d * NTOK + n];
  }
  float* op = outp + (b * QQ + qi) * HH + h * 32;
  for (int d = 0; d < 32; d++) {
    red[t] = o[d]; __syncthreads();
    for (int st = 128; st > 0; st >>= 1) { if (t < st) red[t] += red[t + st]; __syncthreads(); }
    if (t == 0) op[d] = red[0] * inv;
    __syncthreads();
  }
}

// Self attention over 32 query tokens; block per (q,h,b), 32 threads (1 wave).
__global__ void self_attn_kernel(const float* __restrict__ qkv, float* __restrict__ outp)
{
  __shared__ float sv[32];
  int qi = blockIdx.x, h = blockIdx.y, b = blockIdx.z, t = threadIdx.x;
  const float scale = 0.17677669529663689f;
  const float* base = qkv + b * QQ * 768;
  const float* q = base + qi * 768 + h * 32;
  const float* k = base + t * 768 + 256 + h * 32;
  float acc = 0.f;
  #pragma unroll
  for (int d = 0; d < 32; d++) acc += q[d] * k[d];
  sv[t] = acc * scale;
  __syncthreads();
  float mx = -1e30f;
  for (int n = 0; n < 32; n++) mx = fmaxf(mx, sv[n]);
  float sum = 0.f;
  for (int n = 0; n < 32; n++) sum += __expf(sv[n] - mx);
  float inv = 1.f / sum;
  float o = 0.f;   // thread t owns output dim d == t
  for (int n = 0; n < 32; n++)
    o += __expf(sv[n] - mx) * inv * base[n * 768 + 512 + h * 32 + t];
  outp[(b * QQ + qi) * HH + h * 32 + t] = o;
}

// ---------------------------------------------------------------------------
// attn bias: trilinear(sigmoid(prev[S^3])) -> 16^3, then log(max(.,1e-4))
// ---------------------------------------------------------------------------
__global__ void attn_bias_kernel(const float* __restrict__ prev, float* __restrict__ outb,
                                 int S, int total)
{
  int idx = blockIdx.x * 256 + threadIdx.x;
  if (idx >= total) return;
  int x = idx & 15, y = (idx >> 4) & 15, z = (idx >> 8) & 15;
  int bq = idx >> 12;
  const float* p = prev + (long)bq * S * S * S;
  float scale = (float)S / 16.f;
  int i0[3], i1[3]; float w[3]; int c[3] = {z, y, x};
  for (int a = 0; a < 3; a++) {
    float pos = ((float)c[a] + 0.5f) * scale - 0.5f;
    pos = fminf(fmaxf(pos, 0.f), (float)S - 1.f);
    i0[a] = (int)floorf(pos);
    i1[a] = min(i0[a] + 1, S - 1);
    w[a] = pos - (float)i0[a];
  }
  #define SG(zz, yy, xx) sigmoidf(p[((zz) * S + (yy)) * S + (xx)])
  float v =
      (1.f - w[0]) * ((1.f - w[1]) * ((1.f - w[2]) * SG(i0[0], i0[1], i0[2]) + w[2] * SG(i0[0], i0[1], i1[2]))
                    + w[1]         * ((1.f - w[2]) * SG(i0[0], i1[1], i0[2]) + w[2] * SG(i0[0], i1[1], i1[2])))
    + w[0]         * ((1.f - w[1]) * ((1.f - w[2]) * SG(i1[0], i0[1], i0[2]) + w[2] * SG(i1[0], i0[1], i1[2]))
                    + w[1]         * ((1.f - w[2]) * SG(i1[0], i1[1], i0[2]) + w[2] * SG(i1[0], i1[1], i1[2])));
  #undef SG
  outb[idx] = __logf(fmaxf(v, 1e-4f));
}

__global__ void softmax_rows_kernel(const float* __restrict__ in, float* __restrict__ out,
                                    int rows, int O)
{
  int r = blockIdx.x * blockDim.x + threadIdx.x;
  if (r >= rows) return;
  const float* x = in + r * O;
  float mx = -1e30f;
  for (int o = 0; o < O; o++) mx = fmaxf(mx, x[o]);
  float sum = 0.f;
  for (int o = 0; o < O; o++) sum += __expf(x[o] - mx);
  float inv = 1.f / sum;
  for (int o = 0; o < O; o++) out[r * O + o] = __expf(x[o] - mx) * inv;
}

// ---------------------------------------------------------------------------
// Final dense semantic log-probs: out[b][31][N]; cp: [B][32][31] class probs.
// ---------------------------------------------------------------------------
__global__ __launch_bounds__(256) void final_kernel(
    const float* __restrict__ prev, const float* __restrict__ cp,
    float* __restrict__ out, int Npts)
{
  __shared__ float scp[QQ * CCLS];
  int b = blockIdx.y;
  for (int i = threadIdx.x; i < QQ * CCLS; i += 256) {
    int q = i / CCLS, c = i % CCLS;
    scp[i] = cp[(b * QQ + q) * (CCLS + 1) + c];
  }
  __syncthreads();
  int n = blockIdx.x * 256 + threadIdx.x;
  if (n >= Npts) return;
  const float* pb = prev + (long)b * QQ * Npts;
  float fg[CCLS];
  #pragma unroll
  for (int c = 0; c < CCLS; c++) fg[c] = 0.f;
  for (int q = 0; q < QQ; q++) {
    float sgv = sigmoidf(pb[q * Npts + n]);
    const float* cq = &scp[q * CCLS];
    #pragma unroll
    for (int c = 0; c < CCLS; c++) fg[c] += cq[c] * sgv;
  }
  float bg = 1.f;
  #pragma unroll
  for (int c = 0; c < CCLS; c++) bg -= fg[c];
  bg = fmaxf(bg, 1e-6f);
  float tot = bg;
  #pragma unroll
  for (int c = 0; c < CCLS; c++) { fg[c] = fmaxf(fg[c], 1e-6f); tot += fg[c]; }
  float inv = 1.f / fmaxf(tot, 1e-6f);
  float* ob = out + (long)b * (CCLS + 1) * Npts;
  ob[n] = __logf(bg * inv);
  #pragma unroll
  for (int c = 0; c < CCLS; c++) ob[(c + 1) * Npts + n] = __logf(fg[c] * inv);
}

__global__ void bcast_q_kernel(const float* __restrict__ qe, float* __restrict__ qbuf)
{
  int i = blockIdx.x * 256 + threadIdx.x;
  if (i < BB * QQ * HH) qbuf[i] = qe[i % (QQ * HH)];
}

// ---------------------------------------------------------------------------
// Host orchestration
// ---------------------------------------------------------------------------
extern "C" void kernel_launch(void* const* d_in, const int* in_sizes, int n_in,
                              void* d_out, int out_size, void* d_ws, size_t ws_size,
                              hipStream_t stream) {
  (void)in_sizes; (void)n_in; (void)out_size; (void)ws_size;
  auto F = [&](int i) { return (const float*)d_in[i]; };

  // Input order: f0, f1, f2, then params pytree leaves (sorted dict keys):
  //  3 ce_W1  4 ce_W2  5 ce_b1  6 ce_b2  7 ce_ln_b  8 ce_ln_g
  //  9 me_W1 10 me_W2 11 me_b1 12 me_b2 13 me_ln_b 14 me_ln_g
  // 15 query_embed
  // 16+26*s: stage s leaves, sorted:
  //  0 W1   1 W2   2 Wco  3 Wk   4 Wq   5 Wqkv 6 Wso  7 Wv
  //  8 b1   9 b2  10 bco 11 bk  12 bq  13 bqkv 14 bso 15 bv
  // 16 ln1_b 17 ln1_g 18 ln2_b 19 ln2_g 20 ln3_b 21 ln3_g
  // 22 mask_b 23 mask_w 24 token_b 25 token_w
  const float* ce_W1 = F(3);  const float* ce_W2 = F(4);
  const float* ce_b1 = F(5);  const float* ce_b2 = F(6);
  const float* ce_ln_b = F(7); const float* ce_ln_g = F(8);
  const float* me_W1 = F(9);  const float* me_W2 = F(10);
  const float* me_b1 = F(11); const float* me_b2 = F(12);
  const float* me_ln_b = F(13); const float* me_ln_g = F(14);
  const float* query_embed = F(15);

  const int  Cs[3] = {256, 128, 64};
  const int  Ss[3] = {16, 32, 48};
  const long Np[3] = {4096, 32768, 110592};

  // workspace layout (floats)
  float* w = (float*)d_ws;
  float* pooled   = w;               w += (long)BB * 256 * NTOK;   // 2,097,152
  float* tokensT  = w;               w += (long)BB * HH * NTOK;
  float* KT       = w;               w += (long)BB * HH * NTOK;
  float* VT       = w;               w += (long)BB * HH * NTOK;
  float* biasAttn = w;               w += (long)BB * QQ * NTOK;
  float* prev     = w;               w += (long)BB * QQ * 110592;  // 7,077,888
  float* qbuf     = w;               w += BB * QQ * HH;
  float* q1       = w;               w += BB * QQ * HH;
  float* q2       = w;               w += BB * QQ * HH;
  float* tA       = w;               w += BB * QQ * 1024;
  float* tB       = w;               w += BB * QQ * 1024;
  float* qkvB     = w;               w += BB * QQ * 768;
  float* meB      = w;               w += BB * QQ * 256;
  float* ceB      = w;               w += BB * QQ * 32;
  float* EB       = w;               w += BB * QQ * 256;
  float* e0B      = w;               w += BB * QQ;
  float* cpB      = w;               w += BB * QQ * 32;

  const int L = 256;
  auto blocks = [&](long total) { return (int)((total + L - 1) / L); };

  // queries = broadcast(query_embed)
  bcast_q_kernel<<<blocks(BB * QQ * HH), L, 0, stream>>>(query_embed, qbuf);

  for (int s = 0; s < 3; s++) {
    const float* feat = F(s);
    const int base = 16 + 26 * s;
    const float* W1 = F(base + 0),  *W2 = F(base + 1),  *Wco = F(base + 2);
    const float* Wk = F(base + 3),  *Wq = F(base + 4),  *Wqkv = F(base + 5);
    const float* Wso = F(base + 6), *Wv = F(base + 7);
    const float* b1 = F(base + 8),  *b2 = F(base + 9),  *bco = F(base + 10);
    const float* bk = F(base + 11), *bq = F(base + 12), *bqkv = F(base + 13);
    const float* bso = F(base + 14), *bv = F(base + 15);
    const float* ln1_b = F(base + 16), *ln1_g = F(base + 17);
    const float* ln2_b = F(base + 18), *ln2_g = F(base + 19);
    const float* ln3_b = F(base + 20), *ln3_g = F(base + 21);
    const float* mask_b = F(base + 22), *mask_w = F(base + 23);
    const float* token_b = F(base + 24), *token_w = F(base + 25);
    const int C = Cs[s], S = Ss[s];
    const long NP = Np[s];

    // (a) pool feat -> [B][C][4096]  (pool commutes with 1x1x1 conv)
    pool_kernel<<<blocks((long)BB * C * NTOK), L, 0, stream>>>(
        feat, pooled, S, S / 16, BB * C * NTOK);

    // (b) token conv (WMMA): tokensT = token_w[256xC] @ pooled + token_b
    dim3 gTok(NTOK / 256, HH / 32, BB);
    wmma_gemm_bf16<<<gTok, 256, 0, stream>>>(
        token_w, pooled, token_b, tokensT,
        HH, NTOK, C, 0, (long)C * NTOK, 0, (long)HH * NTOK);

    // (c) attention bias from previous stage's prev
    if (s > 0)
      attn_bias_kernel<<<blocks((long)BB * QQ * NTOK), L, 0, stream>>>(
          prev, biasAttn, Ss[s - 1], BB * QQ * NTOK);

    // (d) q projection (tiny)
    linear_kernel<<<blocks(BB * QQ * HH), L, 0, stream>>>(
        qbuf, Wq, bq, tA, BB * QQ, HH, HH, 1, 0);

    // (e,f) K/V projections (WMMA): KT = Wk @ tokensT + bk
    wmma_gemm_bf16<<<gTok, 256, 0, stream>>>(
        Wk, tokensT, bk, KT, HH, NTOK, HH, 0, (long)HH * NTOK, 0, (long)HH * NTOK);
    wmma_gemm_bf16<<<gTok, 256, 0, stream>>>(
        Wv, tokensT, bv, VT, HH, NTOK, HH, 0, (long)HH * NTOK, 0, (long)HH * NTOK);

    // (g) cross attention + out proj + residual LN1
    cross_attn_kernel<<<dim3(QQ, NHD, BB), 256, 0, stream>>>(
        tA, KT, VT, s > 0 ? biasAttn : nullptr, tB);
    linear_kernel<<<blocks(BB * QQ * HH), L, 0, stream>>>(
        tB, Wco, bco, tA, BB * QQ, HH, HH, 1, 0);
    add_ln_kernel<<<BB * QQ, 256, 0, stream>>>(qbuf, tA, ln1_g, ln1_b, q1);

    // (h) self attention + out proj + residual LN2
    linear_kernel<<<blocks(BB * QQ * 768), L, 0, stream>>>(
        q1, Wqkv, bqkv, qkvB, BB * QQ, HH, 768, 1, 0);
    self_attn_kernel<<<dim3(QQ, NHD, BB), 32, 0, stream>>>(qkvB, tA);
    linear_kernel<<<blocks(BB * QQ * HH), L, 0, stream>>>(
        tA, Wso, bso, tB, BB * QQ, HH, HH, 1, 0);
    add_ln_kernel<<<BB * QQ, 256, 0, stream>>>(q1, tB, ln2_g, ln2_b, q2);

    // (i) FFN + residual LN3 -> queries
    linear_kernel<<<blocks(BB * QQ * 1024), L, 0, stream>>>(
        q2, W1, b1, tA, BB * QQ, HH, 1024, 1, 1);         // gelu
    linear_kernel<<<blocks(BB * QQ * HH), L, 0, stream>>>(
        tA, W2, b2, tB, BB * QQ, 1024, HH, 1, 0);
    add_ln_kernel<<<BB * QQ, 256, 0, stream>>>(q2, tB, ln3_g, ln3_b, qbuf);

    // (j) mask / class embedding heads
    add_ln_kernel<<<BB * QQ, 256, 0, stream>>>(qbuf, nullptr, me_ln_g, me_ln_b, q1);
    linear_kernel<<<blocks(BB * QQ * HH), L, 0, stream>>>(
        q1, me_W1, me_b1, tA, BB * QQ, HH, HH, 1, 1);      // gelu
    linear_kernel<<<blocks(BB * QQ * HH), L, 0, stream>>>(
        tA, me_W2, me_b2, meB, BB * QQ, HH, 256, 1, 0);
    add_ln_kernel<<<BB * QQ, 256, 0, stream>>>(qbuf, nullptr, ce_ln_g, ce_ln_b, q1);
    linear_kernel<<<blocks(BB * QQ * HH), L, 0, stream>>>(
        q1, ce_W1, ce_b1, tA, BB * QQ, HH, HH, 1, 1);      // gelu
    linear_kernel<<<blocks(BB * QQ * (CCLS + 1)), L, 0, stream>>>(
        tA, ce_W2, ce_b2, ceB, BB * QQ, HH, CCLS + 1, 1, 0);

    // (k) fold mask path: E = me @ mask_w  [B*Q x C];  e0 = me . mask_b
    linear_kernel<<<blocks(BB * QQ * C), L, 0, stream>>>(
        meB, mask_w, nullptr, EB, BB * QQ, 256, C, 0, 0);
    dot_rows_kernel<<<1, 64, 0, stream>>>(meB, mask_b, e0B, BB * QQ, 256);

    // (l) prev = E @ feat + e0   (WMMA, streams feat exactly once)
    dim3 gPrev((unsigned)(NP / 256), 1, BB);
    wmma_gemm_bf16<<<gPrev, 256, 0, stream>>>(
        EB, feat, e0B, prev, QQ, (int)NP, C,
        (long)QQ * C, (long)C * NP, (long)QQ, (long)QQ * NP);
  }

  // final: class softmax + dense semantic log-probs
  softmax_rows_kernel<<<1, 64, 0, stream>>>(ceB, cpB, BB * QQ, CCLS + 1);
  final_kernel<<<dim3((unsigned)(Np[2] / 256), BB), 256, 0, stream>>>(
      prev, cpB, (float*)d_out, Np[2]);
}
